// ElmanRNN_7078106104269
// MI455X (gfx1250) — matrix operand order repaired and checked by reading, not compiled, not run on hardware
//
#include <hip/hip_runtime.h>
#include <hip/hip_bf16.h>

// ---- problem constants (from reference) ----
#define RB 64      // batch
#define RS 2048    // sequence
#define RI 256     // input dim
#define RH 512     // hidden dim

typedef __attribute__((ext_vector_type(16))) _Float16 v16h;
typedef __attribute__((ext_vector_type(8)))  _Float16 v8h;
typedef __attribute__((ext_vector_type(8)))  float    v8f;
typedef __attribute__((ext_vector_type(4)))  float    v4f;
typedef __attribute__((ext_vector_type(4)))  unsigned int u32x4;
typedef __attribute__((ext_vector_type(8)))  int      i32x8;
typedef __attribute__((ext_vector_type(4)))  int      i32x4;

#if __has_builtin(__builtin_amdgcn_tensor_load_to_lds) && \
    __has_builtin(__builtin_amdgcn_s_wait_tensorcnt)
#define USE_TDM 1
#else
#define USE_TDM 0
#endif

// ---------------------------------------------------------------------------
// Kernel 0: convert W_ih / W_hh to f16 in workspace; fuse b_ih + b_hh (f32).
// ---------------------------------------------------------------------------
__global__ void cvt_weights_kernel(const float* __restrict__ Wih,
                                   const float* __restrict__ Whh,
                                   const float* __restrict__ bih,
                                   const float* __restrict__ bhh,
                                   _Float16* __restrict__ wih16,
                                   _Float16* __restrict__ whh16,
                                   float* __restrict__ bsum) {
    int tid = blockIdx.x * blockDim.x + threadIdx.x;
    if (tid < RH * RI) wih16[tid] = (_Float16)Wih[tid];
    if (tid < RH * RH) whh16[tid] = (_Float16)Whh[tid];
    if (tid < RH)      bsum[tid]  = bih[tid] + bhh[tid];
}

// ---------------------------------------------------------------------------
// Kernel 1: xproj = x @ W_ih^T + (b_ih + b_hh), written IN-PLACE into d_out.
// (B*S = 131072 rows, N = H = 512, K = I = 256). Block = 8 waves, M=16 stripe,
// each wave owns 4 N-tiles. Bandwidth-bound phase (~400 MB @ 23.3 TB/s).
// ---------------------------------------------------------------------------
__global__ __launch_bounds__(256) void xproj_kernel(const float* __restrict__ x,
                                                    const _Float16* __restrict__ wih16,
                                                    const float* __restrict__ bsum,
                                                    float* __restrict__ out) {
    const int lane  = threadIdx.x & 31;
    const int wave  = threadIdx.x >> 5;        // 0..7
    const int row0  = blockIdx.x * 16;         // M tile base over B*S rows
    const int n0    = wave * 64;               // 4 N-tiles of 16 per wave
    const int nlane = lane & 15;
    const int kh    = lane >> 4;               // A/B lane K-half selector

    const float* arow = x + (size_t)(row0 + nlane) * RI;   // A: M = lane&15

    v8f acc[4] = {v8f{}, v8f{}, v8f{}, v8f{}};

    #pragma unroll
    for (int k = 0; k < RI; k += 32) {
        const float* a0 = arow + k + kh * 16;
        v4f x0 = *(const v4f*)(a0);
        v4f x1 = *(const v4f*)(a0 + 4);
        v4f x2 = *(const v4f*)(a0 + 8);
        v4f x3 = *(const v4f*)(a0 + 12);
        v16h afrag;
        #pragma unroll
        for (int j = 0; j < 4; ++j) {
            afrag[j]      = (_Float16)x0[j];
            afrag[4 + j]  = (_Float16)x1[j];
            afrag[8 + j]  = (_Float16)x2[j];
            afrag[12 + j] = (_Float16)x3[j];
        }
        #pragma unroll
        for (int t = 0; t < 4; ++t) {
            const _Float16* bp = wih16 + (size_t)(n0 + t * 16 + nlane) * RI + k + kh * 16;
            v16h bfrag = *(const v16h*)bp;
            acc[t] = __builtin_amdgcn_wmma_f32_16x16x32_f16(
                false, afrag, false, bfrag, (short)0, acc[t], false, false);
        }
    }

    #pragma unroll
    for (int t = 0; t < 4; ++t) {
        const int n  = n0 + t * 16 + nlane;
        const float bv = bsum[n];
        #pragma unroll
        for (int r = 0; r < 8; ++r) {
            const int mm = r + 8 * kh;
            out[(size_t)(row0 + mm) * RH + n] = acc[t][r] + bv;
        }
    }
}

// ---------------------------------------------------------------------------
// TDM helper: issue a tensor_load_to_lds of one xp tile:
//   16 rows (batch rows, stride S*H elements) x 512 contiguous f32 -> LDS.
// D# built per CDNA5 ISA 8.3/8.4: group0 = {count, lds_addr, global_addr,
// type=2}; group1 = {data_size=4B, tensor 512x16, tile 512x16, dim0 stride};
// groups 2/3 zero (2-D tile, tile_dim2..4 = 0). Tracked by TENSORcnt.
// This toolchain exposes the 6-operand builtin (extra zero-filled group).
// ---------------------------------------------------------------------------
#if USE_TDM
__device__ __forceinline__ void tdm_load_xp_tile(unsigned long long gaddr,
                                                 unsigned int lds_off) {
    u32x4 g0;
    g0[0] = 1u;                                        // count=1, user descriptor
    g0[1] = lds_off;                                   // LDS byte address
    g0[2] = (unsigned int)(gaddr & 0xFFFFFFFFull);     // global_addr[31:0]
    g0[3] = (unsigned int)((gaddr >> 32) & 0x1FFFFFFull) | (2u << 30); // [56:32], type=2

    i32x8 g1;
    g1[0] = (int)(2u << 16);                 // workgroup_mask=0, data_size=4B
    g1[1] = (int)((RH & 0xFFFF) << 16);      // tensor_dim0[15:0] = 512
    g1[2] = (int)(((RH >> 16) & 0xFFFF) | (16u << 16)); // dim0 hi | tensor_dim1=16
    g1[3] = (int)((RH & 0xFFFF) << 16);      // tensor_dim1 hi=0 | tile_dim0=512
    g1[4] = 16;                              // tile_dim1=16, tile_dim2=0
    g1[5] = (int)(RS * RH);                  // tensor_dim0_stride lo32 (elements)
    g1[6] = 0;                               // stride hi | tensor_dim1_stride lo
    g1[7] = 0;

    i32x4 z4;
    z4[0] = 0; z4[1] = 0; z4[2] = 0; z4[3] = 0;
    i32x8 z8;
    z8[0] = 0; z8[1] = 0; z8[2] = 0; z8[3] = 0;
    z8[4] = 0; z8[5] = 0; z8[6] = 0; z8[7] = 0;
    __builtin_amdgcn_tensor_load_to_lds(g0, g1, z4, z4, z8, 0);
}
#endif

// ---------------------------------------------------------------------------
// Kernel 2: recurrence. 4 blocks x 1024 threads (32 waves); block g owns batch
// rows [16g,16g+16) -- independent chains, no inter-block sync. h state in LDS
// (f16, 16 KB). Next step's xproj tile is DMA'd into a double-buffered LDS
// stage by the Tensor Data Mover on TENSORcnt, fully overlapping the WMMA
// K-loop (whose waits are LOADcnt-only) -- hides the HBM latency of the
// 268 MB xp stream that cannot all live in L2.
// ---------------------------------------------------------------------------
__global__ __launch_bounds__(1024) void rnn_step_kernel(const _Float16* __restrict__ whh16,
                                                        float* __restrict__ out) {
    __shared__ _Float16 hlds[16 * RH];       // h[m][k]
    __shared__ float    xbuf[2][16 * RH];    // double-buffered xp tiles (64 KB)

    const int lane  = threadIdx.x & 31;
    const int wave  = threadIdx.x >> 5;      // 0..31, one N-tile of 16 each
    const int nlane = lane & 15;
    const int kh    = lane >> 4;
    const int n     = wave * 16 + nlane;
    const int b0    = blockIdx.x * 16;

    // byte address of the xp tile start for step s: &out[(b0*RS + s)*RH]
    const unsigned long long outaddr = (unsigned long long)(uintptr_t)out;
    const unsigned long long tile0   = outaddr + (unsigned long long)b0 * RS * RH * 4ull;

    for (int i = threadIdx.x; i < 16 * RH; i += 1024) hlds[i] = (_Float16)0.f;

#if USE_TDM
    const unsigned int lds_x0 =
        (unsigned int)(uintptr_t)&xbuf[0][0];   // low 32 bits = LDS byte offset
    const unsigned int lds_x1 = (unsigned int)(uintptr_t)&xbuf[1][0];
    if (threadIdx.x < 32) {                      // wave 0 drives the TDM
        tdm_load_xp_tile(tile0, lds_x0);         // xp(s=0) -> xbuf[0]
        __builtin_amdgcn_s_wait_tensorcnt(0);
    }
#endif
    __syncthreads();

    for (int s = 0; s < RS; ++s) {
#if USE_TDM
        // prefetch xp(s+1) via TDM; lands before the end-of-step fence
        if (threadIdx.x < 32 && (s + 1) < RS) {
            tdm_load_xp_tile(tile0 + (unsigned long long)(s + 1) * RH * 4ull,
                             ((s + 1) & 1) ? lds_x1 : lds_x0);
        }
#endif
        // h @ W_hh^T : K = 512 in 16 steps of 32 (A from LDS, B L0/L2-hot)
        v8f acc = {};
        #pragma unroll
        for (int k = 0; k < RH; k += 32) {
            const _Float16* ha = &hlds[nlane * RH + k + kh * 16];
            v8h lo = *(const v8h*)(ha);
            v8h hi = *(const v8h*)(ha + 8);
            v16h afrag;
            #pragma unroll
            for (int j = 0; j < 8; ++j) { afrag[j] = lo[j]; afrag[8 + j] = hi[j]; }

            v16h bfrag = *(const v16h*)(whh16 + (size_t)n * RH + k + kh * 16);

            acc = __builtin_amdgcn_wmma_f32_16x16x32_f16(
                false, afrag, false, bfrag, (short)0, acc, false, false);
        }

        // epilogue: h = tanh(acc + xp) ; xp from the staged LDS buffer
        float hv[8];
        #pragma unroll
        for (int r = 0; r < 8; ++r) {
            const int mm = r + 8 * kh;
#if USE_TDM
            const float xp = xbuf[s & 1][mm * RH + n];
#else
            const float xp = out[((size_t)(b0 + mm) * RS + s) * RH + n];
#endif
            hv[r] = tanhf(acc[r] + xp);
            out[((size_t)(b0 + mm) * RS + s) * RH + n] = hv[r];
        }

        __syncthreads();   // all waves done reading h_{s-1} / xbuf[s&1]
        #pragma unroll
        for (int r = 0; r < 8; ++r) {
            const int mm = r + 8 * kh;
            hlds[mm * RH + n] = (_Float16)hv[r];
        }
#if USE_TDM
        if (threadIdx.x < 32) __builtin_amdgcn_s_wait_tensorcnt(0);
#endif
        __syncthreads();   // h_s and xp(s+1) visible to all waves
    }
}

// ---------------------------------------------------------------------------
extern "C" void kernel_launch(void* const* d_in, const int* in_sizes, int n_in,
                              void* d_out, int out_size, void* d_ws, size_t ws_size,
                              hipStream_t stream) {
    const float* x    = (const float*)d_in[0];   // (B,S,I)
    const float* Wih  = (const float*)d_in[1];   // (H,I)
    const float* Whh  = (const float*)d_in[2];   // (H,H)
    const float* bih  = (const float*)d_in[3];   // (H)
    const float* bhh  = (const float*)d_in[4];   // (H)
    float*       out  = (float*)d_out;           // (B,S,H)

    char* ws = (char*)d_ws;
    _Float16* wih16 = (_Float16*)ws;                                   // 256 KB
    _Float16* whh16 = (_Float16*)(ws + (size_t)RH * RI * 2);           // 512 KB
    float*    bsum  = (float*)(ws + (size_t)RH * RI * 2 + (size_t)RH * RH * 2);

    cvt_weights_kernel<<<(RH * RH + 255) / 256, 256, 0, stream>>>(
        Wih, Whh, bih, bhh, wih16, whh16, bsum);

    xproj_kernel<<<(RB * RS) / 16, 256, 0, stream>>>(x, wih16, bsum, out);

    rnn_step_kernel<<<RB / 16, 1024, 0, stream>>>(whh16, out);
}
